// CrossGraphMessagePassing_45561013075985
// MI455X (gfx1250) — matrix operand order repaired and checked by reading, not compiled, not run on hardware
//
#include <hip/hip_runtime.h>
#include <hip/hip_bf16.h>
#include <math.h>

typedef __attribute__((ext_vector_type(16))) _Float16 v16h;
typedef __attribute__((ext_vector_type(8)))  _Float16 v8h;
typedef __attribute__((ext_vector_type(8)))  float    v8f;

constexpr int   kNP = 100000;
constexpr int   kNL = 40000;
constexpr int   kNE = 2000000;
constexpr int   kH  = 4;
constexpr int   kIN = 150;   // true feature width
constexpr int   kK  = 160;   // K padded to multiple of 32 for 16x16x32 WMMA
constexpr int   kHD = 64;
constexpr int   kLD = 64;
constexpr float kGamma     = 2.0f;          // 1/(2*(CUT/RBF)^2), CUT=8, RBF=16
constexpr float kCStep     = 8.0f / 15.0f;  // linspace(0,8,16) spacing
constexpr float kInvTwoSig2= 1.0f / 32.0f;  // 1/(2*SIG^2), SIG=4

// ---------- helpers ----------

// fast reciprocal: single v_rcp_f32 (TRANS), co-executes with WMMA
__device__ inline float frcp(float x) { return __builtin_amdgcn_rcpf(x); }

// silu via fast rcp instead of IEEE divide sequence
__device__ inline float fsilu(float x) {
  return x * frcp(1.0f + __expf(-x));
}

// order-preserving float<->uint encoding for atomicMax-based segment max
__device__ inline unsigned encf(float f) {
  unsigned u = __float_as_uint(f);
  return (u & 0x80000000u) ? ~u : (u | 0x80000000u);
}
__device__ inline float decf(unsigned u) {
  u = (u & 0x80000000u) ? (u & 0x7FFFFFFFu) : ~u;
  return __uint_as_float(u);
}

// A-fragment for v_wmma_f32_16x16x32_f16: lane<16 holds row m=lane, K 0..15 contiguous;
// lane>=16 holds row m=lane-16, K 16..31. p points at the first 8 halves.
__device__ inline v16h ldfragA(const _Float16* p) {
  v8h lo = *(const v8h*)(p);
  v8h hi = *(const v8h*)(p + 8);
  v16h r;
#pragma unroll
  for (int i = 0; i < 8; ++i) { r[i] = lo[i]; r[i + 8] = hi[i]; }
  return r;
}
// B-fragment: lane<16 holds col n=lane, K 0..15 contiguous; lane>=16 K 16..31.
// Requires weights stored [n][k] (k contiguous).
__device__ inline v16h ldfragB(const _Float16* p) {
  v8h lo = *(const v8h*)(p);
  v8h hi = *(const v8h*)(p + 8);
  v16h r;
#pragma unroll
  for (int i = 0; i < 8; ++i) { r[i] = lo[i]; r[i + 8] = hi[i]; }
  return r;
}

// Build one half (80 features) of the 160-wide padded feature row for edge e.
// khalf=0 writes k 0..79 (h_prot row + h_lig[0..15]), khalf=1 writes k 80..159
// (h_lig[16..63], dist, ux, rbf, dup(ux,uy), zero pad). Returns dist, sets *li_out.
__device__ inline float build_cat_half(_Float16* crow, int khalf,
                                       const float* __restrict__ h_prot,
                                       const float* __restrict__ h_lig,
                                       const float* __restrict__ prot_pos,
                                       const float* __restrict__ lig_pos,
                                       const int* __restrict__ ce, int e, int* li_out) {
  int pi = ce[e];
  int li = ce[kNE + e];
  *li_out = li;
  float dx = lig_pos[li * 3 + 0] - prot_pos[pi * 3 + 0];
  float dy = lig_pos[li * 3 + 1] - prot_pos[pi * 3 + 1];
  float dz = lig_pos[li * 3 + 2] - prot_pos[pi * 3 + 2];
  float dist = sqrtf(dx * dx + dy * dy + dz * dz);
  float inv = frcp(dist + 1e-8f);
  float ux = dx * inv, uy = dy * inv, uz = dz * inv;
  if (khalf == 0) {
    const float4* hp4 = (const float4*)(h_prot + (long)pi * 64);
#pragma unroll
    for (int q = 0; q < 16; ++q) {
      float4 v = hp4[q];
      crow[q * 4 + 0] = (_Float16)v.x; crow[q * 4 + 1] = (_Float16)v.y;
      crow[q * 4 + 2] = (_Float16)v.z; crow[q * 4 + 3] = (_Float16)v.w;
    }
    const float4* hl4 = (const float4*)(h_lig + (long)li * 64);
#pragma unroll
    for (int q = 0; q < 4; ++q) {
      float4 v = hl4[q];
      crow[64 + q * 4 + 0] = (_Float16)v.x; crow[64 + q * 4 + 1] = (_Float16)v.y;
      crow[64 + q * 4 + 2] = (_Float16)v.z; crow[64 + q * 4 + 3] = (_Float16)v.w;
    }
  } else {
    const float4* hl4 = (const float4*)(h_lig + (long)li * 64);
#pragma unroll
    for (int q = 4; q < 16; ++q) {
      float4 v = hl4[q];
      crow[64 + q * 4 + 0] = (_Float16)v.x; crow[64 + q * 4 + 1] = (_Float16)v.y;
      crow[64 + q * 4 + 2] = (_Float16)v.z; crow[64 + q * 4 + 3] = (_Float16)v.w;
    }
    crow[128] = (_Float16)dist;
    crow[129] = (_Float16)ux; crow[130] = (_Float16)uy; crow[131] = (_Float16)uz;
#pragma unroll
    for (int j = 0; j < 16; ++j) {
      float c = kCStep * j;
      float d = dist - c;
      crow[132 + j] = (_Float16)__expf(-kGamma * d * d);
    }
    crow[148] = (_Float16)ux; crow[149] = (_Float16)uy;
#pragma unroll
    for (int k = kIN; k < kK; ++k) crow[k] = (_Float16)0.0f;
  }
  return dist;
}

// ---------- pass A: per-edge attention logits (WMMA) ----------
__global__ void __launch_bounds__(128)
cgmp_logits(const float* __restrict__ h_prot, const float* __restrict__ h_lig,
            const float* __restrict__ prot_pos, const float* __restrict__ lig_pos,
            const int* __restrict__ ce,
            const float* __restrict__ W1a, const float* __restrict__ b1a,
            const float* __restrict__ W2a, const float* __restrict__ b2a,
            float* __restrict__ logits) {
  __shared__ alignas(16) _Float16 sW1[kH * kHD * kK];   // W1a^T as [h][n][k], k padded/zeroed
  __shared__ float sB1[kH * kHD];
  __shared__ float sW2[kH * kHD];
  __shared__ float sB2[kH];
  __shared__ alignas(16) _Float16 sCat[4 * 16 * kK];    // per-wave 16x160 f16 tile

  for (int idx = threadIdx.x; idx < kH * kHD * kK; idx += blockDim.x) {
    int h = idx / (kHD * kK);
    int n = (idx / kK) % kHD;
    int k = idx % kK;
    float w = (k < kIN) ? W1a[(h * kIN + k) * kHD + n] : 0.0f;
    sW1[idx] = (_Float16)w;
  }
  for (int idx = threadIdx.x; idx < kH * kHD; idx += blockDim.x) {
    sB1[idx] = b1a[idx];
    sW2[idx] = W2a[idx];
  }
  if (threadIdx.x < kH) sB2[threadIdx.x] = b2a[threadIdx.x];
  __syncthreads();

  const int lane  = threadIdx.x & 31;
  const int wave  = threadIdx.x >> 5;
  const int eL    = lane & 15;
  const int khalf = lane >> 4;
  const int tile  = blockIdx.x * 4 + wave;
  const int e0    = tile * 16;

  _Float16* crow = sCat + wave * (16 * kK) + eL * kK;
  int li;
  build_cat_half(crow, khalf, h_prot, h_lig, prot_pos, lig_pos, ce, e0 + eL, &li);
  asm volatile("s_wait_dscnt 0" ::: "memory");  // per-wave LDS fill visible

#pragma unroll
  for (int h = 0; h < kH; ++h) {
    v8f acc[4];
#pragma unroll
    for (int nt = 0; nt < 4; ++nt)
#pragma unroll
      for (int r = 0; r < 8; ++r) acc[nt][r] = 0.0f;

#pragma unroll
    for (int kb = 0; kb < 5; ++kb) {
      v16h a = ldfragA(sCat + wave * (16 * kK) + eL * kK + kb * 32 + khalf * 16);
#pragma unroll
      for (int nt = 0; nt < 4; ++nt) {
        int n = (lane & 15) + nt * 16;
        v16h b = ldfragB(sW1 + (h * kHD + n) * kK + kb * 32 + khalf * 16);
        acc[nt] = __builtin_amdgcn_wmma_f32_16x16x32_f16(
            false, a, false, b, (short)0, acc[nt], false, false);
      }
    }
    // bias + silu, then dot with W2a column; C layout: VGPR r = row (r + 8*khalf)
    float p[8];
#pragma unroll
    for (int r = 0; r < 8; ++r) p[r] = 0.0f;
#pragma unroll
    for (int nt = 0; nt < 4; ++nt) {
      int n = (lane & 15) + nt * 16;
      float b1 = sB1[h * kHD + n];
      float w2 = sW2[h * kHD + n];
#pragma unroll
      for (int r = 0; r < 8; ++r) {
        float x = acc[nt][r] + b1;
        p[r] += fsilu(x) * w2;
      }
    }
#pragma unroll
    for (int r = 0; r < 8; ++r) {
#pragma unroll
      for (int off = 1; off < 16; off <<= 1) p[r] += __shfl_xor(p[r], off, 32);
    }
    if (eL == 0) {
      int mb = khalf * 8;
#pragma unroll
      for (int r = 0; r < 8; ++r)
        logits[(long)(e0 + mb + r) * kH + h] = p[r] + sB2[h];
    }
  }
}

// ---------- small passes ----------
__global__ void cgmp_init(unsigned* __restrict__ segmax, float* __restrict__ den,
                          float* __restrict__ agg) {
  int i = blockIdx.x * blockDim.x + threadIdx.x;
  if (i < kNL * kLD) agg[i] = 0.0f;
  if (i < kNL * kH) { segmax[i] = 0u; den[i] = 0.0f; }
}

__global__ void cgmp_segmax(const float* __restrict__ logits, const int* __restrict__ ce,
                            unsigned* __restrict__ segmax) {
  int idx = blockIdx.x * blockDim.x + threadIdx.x;
  if (idx >= kNE * kH) return;
  int e = idx >> 2, h = idx & 3;
  int li = ce[kNE + e];
  atomicMax(&segmax[li * kH + h], encf(logits[idx]));
}

__global__ void cgmp_den(const float* __restrict__ logits, const int* __restrict__ ce,
                         const unsigned* __restrict__ segmax, float* __restrict__ den) {
  int idx = blockIdx.x * blockDim.x + threadIdx.x;
  if (idx >= kNE * kH) return;
  int e = idx >> 2, h = idx & 3;
  int li = ce[kNE + e];
  float m = decf(segmax[li * kH + h]);
  atomicAdd(&den[li * kH + h], __expf(logits[idx] - m));
}

// ---------- pass B: values + softmax-weighted scatter (2x WMMA) ----------
__global__ void __launch_bounds__(128)
cgmp_agg(const float* __restrict__ h_prot, const float* __restrict__ h_lig,
         const float* __restrict__ prot_pos, const float* __restrict__ lig_pos,
         const int* __restrict__ ce,
         const float* __restrict__ W1p, const float* __restrict__ b1p,
         const float* __restrict__ W2p, const float* __restrict__ b2p,
         const float* __restrict__ logits, const unsigned* __restrict__ segmax,
         const float* __restrict__ den, float* __restrict__ agg) {
  __shared__ alignas(16) _Float16 sW1[kH * kHD * kK];   // W1p^T [h][n][k160]
  __shared__ alignas(16) _Float16 sW2[kH * kLD * kHD];  // W2p^T [h][n][k64]
  __shared__ float sB1[kH * kHD];
  __shared__ float sB2[kH * kLD];
  __shared__ alignas(16) _Float16 sCat[4 * 16 * kK];
  __shared__ alignas(16) _Float16 sS[4 * 16 * kHD];     // silu(GEMM1) staged as A operand

  for (int idx = threadIdx.x; idx < kH * kHD * kK; idx += blockDim.x) {
    int h = idx / (kHD * kK);
    int n = (idx / kK) % kHD;
    int k = idx % kK;
    float w = (k < kIN) ? W1p[(h * kIN + k) * kHD + n] : 0.0f;
    sW1[idx] = (_Float16)w;
  }
  for (int idx = threadIdx.x; idx < kH * kLD * kHD; idx += blockDim.x) {
    int h = idx / (kLD * kHD);
    int n = (idx / kHD) % kLD;
    int k = idx % kHD;
    sW2[idx] = (_Float16)W2p[(h * kHD + k) * kLD + n];
  }
  for (int idx = threadIdx.x; idx < kH * kHD; idx += blockDim.x) sB1[idx] = b1p[idx];
  for (int idx = threadIdx.x; idx < kH * kLD; idx += blockDim.x) sB2[idx] = b2p[idx];
  __syncthreads();

  const int lane  = threadIdx.x & 31;
  const int wave  = threadIdx.x >> 5;
  const int eL    = lane & 15;
  const int khalf = lane >> 4;
  const int tile  = blockIdx.x * 4 + wave;
  const int e0    = tile * 16;

  _Float16* crow = sCat + wave * (16 * kK) + eL * kK;
  int li;
  float dist = build_cat_half(crow, khalf, h_prot, h_lig, prot_pos, lig_pos, ce, e0 + eL, &li);
  float decay = __expf(-dist * dist * kInvTwoSig2);
  asm volatile("s_wait_dscnt 0" ::: "memory");

  float outAcc[4][8];
#pragma unroll
  for (int nt = 0; nt < 4; ++nt)
#pragma unroll
    for (int r = 0; r < 8; ++r) outAcc[nt][r] = 0.0f;

  _Float16* srow = sS + wave * (16 * kHD);

#pragma unroll
  for (int h = 0; h < kH; ++h) {
    // GEMM1: cat (16x160) @ W1p (160x64)
    v8f acc[4];
#pragma unroll
    for (int nt = 0; nt < 4; ++nt)
#pragma unroll
      for (int r = 0; r < 8; ++r) acc[nt][r] = 0.0f;
#pragma unroll
    for (int kb = 0; kb < 5; ++kb) {
      v16h a = ldfragA(sCat + wave * (16 * kK) + eL * kK + kb * 32 + khalf * 16);
#pragma unroll
      for (int nt = 0; nt < 4; ++nt) {
        int n = (lane & 15) + nt * 16;
        v16h b = ldfragB(sW1 + (h * kHD + n) * kK + kb * 32 + khalf * 16);
        acc[nt] = __builtin_amdgcn_wmma_f32_16x16x32_f16(
            false, a, false, b, (short)0, acc[nt], false, false);
      }
    }
    // bias + silu -> LDS (C layout -> row-major f16), to refeed as A operand
#pragma unroll
    for (int nt = 0; nt < 4; ++nt) {
      int n = (lane & 15) + nt * 16;
      float b1 = sB1[h * kHD + n];
#pragma unroll
      for (int r = 0; r < 8; ++r) {
        float x = acc[nt][r] + b1;
        srow[(r + 8 * khalf) * kHD + n] = (_Float16)fsilu(x);
      }
    }
    asm volatile("s_wait_dscnt 0" ::: "memory");

    // GEMM2: S (16x64) @ W2p (64x64)
    v8f t[4];
#pragma unroll
    for (int nt = 0; nt < 4; ++nt)
#pragma unroll
      for (int r = 0; r < 8; ++r) t[nt][r] = 0.0f;
#pragma unroll
    for (int kb = 0; kb < 2; ++kb) {
      v16h a = ldfragA(srow + eL * kHD + kb * 32 + khalf * 16);
#pragma unroll
      for (int nt = 0; nt < 4; ++nt) {
        int n = (lane & 15) + nt * 16;
        v16h b = ldfragB(sW2 + (h * kLD + n) * kHD + kb * 32 + khalf * 16);
        t[nt] = __builtin_amdgcn_wmma_f32_16x16x32_f16(
            false, a, false, b, (short)0, t[nt], false, false);
      }
    }
    // alpha = exp(logit - segmax)/(den + 1e-9) * decay, broadcast per row via shfl
    float lg = logits[(long)(e0 + eL) * kH + h];
    float mx = decf(segmax[li * kH + h]);
    float dn = den[li * kH + h];
    float alpha = __expf(lg - mx) * frcp(dn + 1e-9f) * decay;
    float arow[8];
#pragma unroll
    for (int r = 0; r < 8; ++r) arow[r] = __shfl(alpha, r + 8 * khalf, 32);
#pragma unroll
    for (int nt = 0; nt < 4; ++nt) {
      int n = (lane & 15) + nt * 16;
      float b2 = sB2[h * kLD + n];
#pragma unroll
      for (int r = 0; r < 8; ++r) outAcc[nt][r] += arow[r] * (t[nt][r] + b2);
    }
  }

  // scatter-add rows into agg[l_idx]
  int lrow[8];
#pragma unroll
  for (int r = 0; r < 8; ++r) lrow[r] = __shfl(li, r + 8 * khalf, 32);
#pragma unroll
  for (int nt = 0; nt < 4; ++nt) {
    int n = (lane & 15) + nt * 16;
#pragma unroll
    for (int r = 0; r < 8; ++r)
      atomicAdd(&agg[(long)lrow[r] * kLD + n], outAcc[nt][r]);
  }
}

// ---------- LayerNorm over h_lig + agg ----------
__global__ void __launch_bounds__(128)
cgmp_ln(const float* __restrict__ h_lig, const float* __restrict__ agg,
        const float* __restrict__ ln_w, const float* __restrict__ ln_b,
        float* __restrict__ out) {
  int lane = threadIdx.x & 31;
  int wave = threadIdx.x >> 5;
  int row = blockIdx.x * 4 + wave;
  if (row >= kNL) return;
  float x0 = h_lig[(long)row * 64 + lane]      + agg[(long)row * 64 + lane];
  float x1 = h_lig[(long)row * 64 + 32 + lane] + agg[(long)row * 64 + 32 + lane];
  float s = x0 + x1, q = x0 * x0 + x1 * x1;
#pragma unroll
  for (int off = 1; off < 32; off <<= 1) {
    s += __shfl_xor(s, off, 32);
    q += __shfl_xor(q, off, 32);
  }
  float mean = s * (1.0f / 64.0f);
  float var  = q * (1.0f / 64.0f) - mean * mean;
  float rs = rsqrtf(var + 1e-5f);
  out[(long)row * 64 + lane]      = (x0 - mean) * rs * ln_w[lane]      + ln_b[lane];
  out[(long)row * 64 + 32 + lane] = (x1 - mean) * rs * ln_w[32 + lane] + ln_b[32 + lane];
}

// ---------- launcher ----------
extern "C" void kernel_launch(void* const* d_in, const int* in_sizes, int n_in,
                              void* d_out, int out_size, void* d_ws, size_t ws_size,
                              hipStream_t stream) {
  const float* h_prot   = (const float*)d_in[0];
  const float* h_lig    = (const float*)d_in[1];
  const float* prot_pos = (const float*)d_in[2];
  const float* lig_pos  = (const float*)d_in[3];
  const int*   ce       = (const int*)d_in[4];
  const float* W1a = (const float*)d_in[5];
  const float* b1a = (const float*)d_in[6];
  const float* W2a = (const float*)d_in[7];
  const float* b2a = (const float*)d_in[8];
  const float* W1p = (const float*)d_in[9];
  const float* b1p = (const float*)d_in[10];
  const float* W2p = (const float*)d_in[11];
  const float* b2p = (const float*)d_in[12];
  const float* ln_w = (const float*)d_in[13];
  const float* ln_b = (const float*)d_in[14];

  // workspace layout (bytes):
  //   logits : E*H*4           = 32,000,000
  //   segmax : NL*H*4          =    640,000
  //   den    : NL*H*4          =    640,000
  //   agg    : NL*64*4         = 10,240,000   (total ~43.5 MB)
  char* ws = (char*)d_ws;
  float*    logits = (float*)(ws);
  unsigned* segmax = (unsigned*)(ws + 32000000);
  float*    den    = (float*)(ws + 32640000);
  float*    agg    = (float*)(ws + 33280000);

  float* out_prot = (float*)d_out;
  float* out_lig  = (float*)d_out + (long)kNP * 64;

  // passthrough h_prot output
  hipMemcpyAsync(out_prot, d_in[0], (size_t)kNP * 64 * sizeof(float),
                 hipMemcpyDeviceToDevice, stream);

  cgmp_init<<<(kNL * kLD + 255) / 256, 256, 0, stream>>>(segmax, den, agg);

  const int tileBlocks = kNE / 16 / 4;  // 31250 blocks, 4 waves/block, 16 edges/wave
  cgmp_logits<<<tileBlocks, 128, 0, stream>>>(h_prot, h_lig, prot_pos, lig_pos, ce,
                                              W1a, b1a, W2a, b2a, logits);
  cgmp_segmax<<<(kNE * kH + 255) / 256, 256, 0, stream>>>(logits, ce, segmax);
  cgmp_den<<<(kNE * kH + 255) / 256, 256, 0, stream>>>(logits, ce, segmax, den);
  cgmp_agg<<<tileBlocks, 128, 0, stream>>>(h_prot, h_lig, prot_pos, lig_pos, ce,
                                           W1p, b1p, W2p, b2p, logits, segmax, den, agg);
  cgmp_ln<<<(kNL + 3) / 4, 128, 0, stream>>>(h_lig, agg, ln_w, ln_b, out_lig);
}
